// Decoder_14645838479874
// MI455X (gfx1250) — matrix operand order repaired and checked by reading, not compiled, not run on hardware
//
#include <hip/hip_runtime.h>
#include <hip/hip_bf16.h>

// Dims fixed by setup_inputs()
#define S_LEN 512
#define B_DIM 256
#define E_DIM 512
#define H_DIM 512
#define D_DIM 64
#define P_LEN 20
#define T_STEPS 100   // real_poses_len (device scalar; must be host-known for graph capture)
#define G3 (3 * H_DIM)

typedef __attribute__((ext_vector_type(16))) __bf16 v16bf;
typedef __attribute__((ext_vector_type(8)))  __bf16 v8bf;
typedef __attribute__((ext_vector_type(8)))  float  v8f;
typedef __attribute__((ext_vector_type(4)))  float  v4f;

// ---- WMMA tile loaders (CDNA5 16x16x32 bf16 layouts) -----------------------
// A 16x32: lane holds row (lane&15); K chunks [c..c+7] and [c+16..c+23], c=(lane>>4)*8
__device__ __forceinline__ v16bf load_a_bf16(const __bf16* rowptr, int k0, int lane) {
    int c = (lane >> 4) << 3;
    v8bf lo = *(const v8bf*)(rowptr + k0 + c);
    v8bf hi = *(const v8bf*)(rowptr + k0 + c + 16);
    v16bf r;
#pragma unroll
    for (int i = 0; i < 8; ++i) { r[i] = lo[i]; r[i + 8] = hi[i]; }
    return r;
}
// A from fp32 source (convert inline)
__device__ __forceinline__ v16bf load_a_f32(const float* rowptr, int k0, int lane) {
    int c = (lane >> 4) << 3;
    const v4f* p = (const v4f*)(rowptr + k0 + c);
    const v4f* q = (const v4f*)(rowptr + k0 + c + 16);
    v4f a0 = p[0], a1 = p[1], b0 = q[0], b1 = q[1];
    v16bf r;
#pragma unroll
    for (int i = 0; i < 4; ++i) {
        r[i] = (__bf16)a0[i]; r[4 + i] = (__bf16)a1[i];
        r[8 + i] = (__bf16)b0[i]; r[12 + i] = (__bf16)b1[i];
    }
    return r;
}
// B 32x16: lane holds column (lane&15); K chunks [c..c+7] and [c+16..c+23], c=(lane>>4)*8
__device__ __forceinline__ v16bf load_b_bf16(const __bf16* colptr, int k0, int lane) {
    int c = (lane >> 4) << 3;
    v8bf lo = *(const v8bf*)(colptr + k0 + c);
    v8bf hi = *(const v8bf*)(colptr + k0 + c + 16);
    v16bf r;
#pragma unroll
    for (int i = 0; i < 8; ++i) { r[i] = lo[i]; r[i + 8] = hi[i]; }
    return r;
}

__device__ __forceinline__ v8f wmma_bf16(v16bf a, v16bf b, v8f c) {
    return __builtin_amdgcn_wmma_f32_16x16x32_bf16(false, a, false, b, (short)0, c, false, false);
}

// ---- small conversion kernels ---------------------------------------------
__global__ void cvt_f32_bf16(const float* __restrict__ src, __bf16* __restrict__ dst, int n) {
    int i = blockIdx.x * blockDim.x + threadIdx.x;
    if (i < n) dst[i] = (__bf16)src[i];
}

// build initial-hidden A matrix: A[b][e] = encoder_hidden[2 + e/256][b][e%256]
__global__ void build_init_A(const float* __restrict__ ehid, __bf16* __restrict__ Abf) {
    int i = blockIdx.x * blockDim.x + threadIdx.x;
    if (i >= B_DIM * E_DIM) return;
    int b = i >> 9, e = i & 511;
    float v = ehid[(size_t)(2 + (e >> 8)) * B_DIM * 256 + (size_t)b * 256 + (e & 255)];
    Abf[i] = (__bf16)v;
}

// ---- generic WMMA GEMM: C = A(MxK,bf16) @ W(NxK,bf16)^T + bias ------------
// Each wave computes a 16(M)x64(N) strip (A tile register-reused over 4 B tiles).
// block = 4 waves stacked along M; grid = (N/64, M/64)
__global__ void gemm_bf16_wmma(const __bf16* __restrict__ A, int lda,
                               const __bf16* __restrict__ W, int ldw,
                               const float* __restrict__ bias,
                               float* __restrict__ Cf, int ldc,
                               __bf16* __restrict__ Cb, int ldcb,
                               int M, int N, int K) {
    int lane = threadIdx.x & 31;
    int wave = threadIdx.x >> 5;
    int n0 = blockIdx.x * 64;
    int m0 = (blockIdx.y * 4 + wave) * 16;
    if (n0 >= N || m0 >= M) return;   // wave-uniform -> EXEC stays all-1
    const __bf16* Ap = A + (size_t)(m0 + (lane & 15)) * lda;
    const __bf16* Wp = W + (size_t)(n0 + (lane & 15)) * ldw;
    v8f acc[4] = {{}, {}, {}, {}};
    for (int k = 0; k < K; k += 32) {
        v16bf a = load_a_bf16(Ap, k, lane);
#pragma unroll
        for (int j = 0; j < 4; ++j) {
            v16bf b = load_b_bf16(Wp + (size_t)j * 16 * ldw, k, lane);
            acc[j] = wmma_bf16(a, b, acc[j]);
        }
    }
    int half = lane >> 4;
#pragma unroll
    for (int j = 0; j < 4; ++j) {
        int col = n0 + j * 16 + (lane & 15);
        float bv = bias ? bias[col] : 0.f;
#pragma unroll
        for (int r = 0; r < 8; ++r) {
            int mrow = m0 + r + half * 8;      // C/D layout: VGPR r -> M=r / M=8+r
            float v = acc[j][r] + bv;
            if (Cf) Cf[(size_t)mrow * ldc + col] = v;
            if (Cb) Cb[(size_t)mrow * ldcb + col] = (__bf16)v;
        }
    }
}

// ---- enc_proj: ep[b][s][h] = sum_e enc[s][b][e]*W_att[h][e] + b_att[h] (bf16 out)
__global__ void enc_proj_wmma(const float* __restrict__ enc,
                              const __bf16* __restrict__ Watt,
                              const float* __restrict__ batt,
                              __bf16* __restrict__ ep) {
    int lane = threadIdx.x & 31;
    int wave = threadIdx.x >> 5;
    int n0 = blockIdx.x * 64;                 // grid.x = H/64 = 8
    int m0 = (blockIdx.y * 4 + wave) * 16;    // grid.y = S*B/64 = 2048
    const float*  Ap = enc + (size_t)(m0 + (lane & 15)) * E_DIM;
    const __bf16* Wp = Watt + (size_t)(n0 + (lane & 15)) * E_DIM;
    v8f acc[4] = {{}, {}, {}, {}};
    for (int k = 0; k < E_DIM; k += 32) {
        v16bf a = load_a_f32(Ap, k, lane);
#pragma unroll
        for (int j = 0; j < 4; ++j) {
            v16bf b = load_b_bf16(Wp + (size_t)j * 16 * E_DIM, k, lane);
            acc[j] = wmma_bf16(a, b, acc[j]);
        }
    }
    int half = lane >> 4;
#pragma unroll
    for (int j = 0; j < 4; ++j) {
        int col = n0 + j * 16 + (lane & 15);
        float bv = batt[col];
#pragma unroll
        for (int r = 0; r < 8; ++r) {
            int m = m0 + r + half * 8;        // m = s*B + b
            int s = m >> 8;                   // / B_DIM
            int b = m & (B_DIM - 1);
            ep[((size_t)b * S_LEN + s) * H_DIM + col] = (__bf16)(acc[j][r] + bv);
        }
    }
}

// ---- fused GRU gate GEMM: gi = x@W_ih^T (K=64), gh = h@W_hh^T (K=512) ------
// grid = (1536/64 = 24, 256/64 = 4), block 128
__global__ void gru_gemm_wmma(const __bf16* __restrict__ xbf,
                              const __bf16* __restrict__ hbf,
                              const __bf16* __restrict__ Wih,
                              const __bf16* __restrict__ Whh,
                              float* __restrict__ gi, float* __restrict__ gh) {
    int lane = threadIdx.x & 31;
    int wave = threadIdx.x >> 5;
    int n0 = blockIdx.x * 64;
    int m0 = (blockIdx.y * 4 + wave) * 16;
    const __bf16* Ax = xbf + (size_t)(m0 + (lane & 15)) * D_DIM;
    const __bf16* Wx = Wih + (size_t)(n0 + (lane & 15)) * D_DIM;
    const __bf16* Ah = hbf + (size_t)(m0 + (lane & 15)) * H_DIM;
    const __bf16* Wh = Whh + (size_t)(n0 + (lane & 15)) * H_DIM;
    v8f ai[4] = {{}, {}, {}, {}};
#pragma unroll
    for (int k = 0; k < D_DIM; k += 32) {
        v16bf a = load_a_bf16(Ax, k, lane);
#pragma unroll
        for (int j = 0; j < 4; ++j) {
            v16bf b = load_b_bf16(Wx + (size_t)j * 16 * D_DIM, k, lane);
            ai[j] = wmma_bf16(a, b, ai[j]);
        }
    }
    v8f ah[4] = {{}, {}, {}, {}};
    for (int k = 0; k < H_DIM; k += 32) {
        v16bf a = load_a_bf16(Ah, k, lane);
#pragma unroll
        for (int j = 0; j < 4; ++j) {
            v16bf b = load_b_bf16(Wh + (size_t)j * 16 * H_DIM, k, lane);
            ah[j] = wmma_bf16(a, b, ah[j]);
        }
    }
    int half = lane >> 4;
#pragma unroll
    for (int j = 0; j < 4; ++j) {
        int col = n0 + j * 16 + (lane & 15);
#pragma unroll
        for (int r = 0; r < 8; ++r) {
            int mrow = m0 + r + half * 8;
            gi[(size_t)mrow * G3 + col] = ai[j][r];
            gh[(size_t)mrow * G3 + col] = ah[j][r];
        }
    }
}

// ---- GRU elementwise gates (biases folded here; n-gate needs gh separate) --
__global__ void gru_gates(const float* __restrict__ gi, const float* __restrict__ gh,
                          const float* __restrict__ bih, const float* __restrict__ bhh,
                          float* __restrict__ h, __bf16* __restrict__ hbf,
                          __bf16* __restrict__ concatbf) {
    int i = blockIdx.x * blockDim.x + threadIdx.x;   // B*H
    if (i >= B_DIM * H_DIM) return;
    int b = i >> 9, j = i & 511;
    const float* gib = gi + (size_t)b * G3;
    const float* ghb = gh + (size_t)b * G3;
    float ir = gib[j] + bih[j],               hr = ghb[j] + bhh[j];
    float iz = gib[512 + j] + bih[512 + j],   hz = ghb[512 + j] + bhh[512 + j];
    float in = gib[1024 + j] + bih[1024 + j], hn = ghb[1024 + j] + bhh[1024 + j];
    float r = 1.f / (1.f + __expf(-(ir + hr)));
    float z = 1.f / (1.f + __expf(-(iz + hz)));
    float n = tanhf(in + r * hn);
    float hnew = (1.f - z) * n + z * h[i];
    h[i] = hnew;
    hbf[i] = (__bf16)hnew;
    concatbf[(size_t)b * 1024 + j] = (__bf16)hnew;   // first half of [hc|ctx]
}

// ---- attention: one block per batch, single-pass online softmax ------------
// Each wave owns s = wave, wave+8, ...; lane owns h chunk [lane*16, lane*16+16).
// ep row is read ONCE per step (flash-style), halving the dominant L2 traffic.
__global__ void attention_kernel(const __bf16* __restrict__ ep,
                                 const float* __restrict__ h,
                                 __bf16* __restrict__ concatbf) {
    __shared__ float ctx[H_DIM];
    __shared__ float wm[8], wl[8];
    int b = blockIdx.x;
    int tid = threadIdx.x;                 // 256 threads = 8 waves
    int lane = tid & 31, wave = tid >> 5;
    ctx[tid] = 0.f; ctx[tid + 256] = 0.f;
    float hc[16];
    const float* hb = h + (size_t)b * H_DIM + lane * 16;
#pragma unroll
    for (int i = 0; i < 16; ++i) hc[i] = hb[i];
    __syncthreads();
    const __bf16* epb = ep + (size_t)b * S_LEN * H_DIM;
    float m = -3.0e38f, l = 0.f;
    float acc[16];
#pragma unroll
    for (int i = 0; i < 16; ++i) acc[i] = 0.f;
    for (int s = wave; s < S_LEN; s += 8) {
        const __bf16* rowp = epb + (size_t)s * H_DIM + lane * 16;
        v8bf r0 = *(const v8bf*)rowp;
        v8bf r1 = *(const v8bf*)(rowp + 8);
        float row[16];
#pragma unroll
        for (int i = 0; i < 8; ++i) { row[i] = (float)r0[i]; row[8 + i] = (float)r1[i]; }
        float d = 0.f;
#pragma unroll
        for (int i = 0; i < 16; ++i) d += row[i] * hc[i];
        for (int off = 16; off > 0; off >>= 1) d += __shfl_xor(d, off);
        float mn = fmaxf(m, d);
        float sc = __expf(m - mn);
        float w  = __expf(d - mn);
        l = l * sc + w;
#pragma unroll
        for (int i = 0; i < 16; ++i) acc[i] = acc[i] * sc + w * row[i];
        m = mn;
    }
    if (lane == 0) { wm[wave] = m; wl[wave] = l; }
    __syncthreads();
    float gm = wm[0];
#pragma unroll
    for (int i = 1; i < 8; ++i) gm = fmaxf(gm, wm[i]);
    float lt = 0.f;
#pragma unroll
    for (int i = 0; i < 8; ++i) lt += wl[i] * __expf(wm[i] - gm);
    float f = __expf(m - gm);
#pragma unroll
    for (int i = 0; i < 16; ++i) atomicAdd(&ctx[lane * 16 + i], acc[i] * f);
    __syncthreads();
    float inv = 1.f / lt;
    concatbf[(size_t)b * 1024 + 512 + tid] = (__bf16)(ctx[tid] * inv);
    concatbf[(size_t)b * 1024 + 768 + tid] = (__bf16)(ctx[tid + 256] * inv);
}

extern "C" void kernel_launch(void* const* d_in, const int* in_sizes, int n_in,
                              void* d_out, int out_size, void* d_ws, size_t ws_size,
                              hipStream_t stream) {
    const float* enc_states = (const float*)d_in[0];
    const float* enc_hidden = (const float*)d_in[1];
    const float* prev_poses = (const float*)d_in[2];
    const float* W_enc_dec  = (const float*)d_in[3];
    const float* b_enc_dec  = (const float*)d_in[4];
    const float* W_ih       = (const float*)d_in[5];
    const float* W_hh       = (const float*)d_in[6];
    const float* b_ih       = (const float*)d_in[7];
    const float* b_hh       = (const float*)d_in[8];
    const float* W_att      = (const float*)d_in[9];
    const float* b_att      = (const float*)d_in[10];
    const float* W_out      = (const float*)d_in[11];
    const float* b_out      = (const float*)d_in[12];
    float* out = (float*)d_out;

    // workspace carve-up (~140 MB; dominated by bf16 enc_proj = 128 MB, L2-resident)
    char* ws = (char*)d_ws;
    size_t off = 0;
    auto carve = [&](size_t bytes) -> char* {
        char* p = ws + off; off += (bytes + 255) & ~(size_t)255; return p;
    };
    __bf16* ep        = (__bf16*)carve((size_t)B_DIM * S_LEN * H_DIM * 2);
    __bf16* Wih_bf    = (__bf16*)carve((size_t)G3 * D_DIM * 2);
    __bf16* Whh_bf    = (__bf16*)carve((size_t)G3 * H_DIM * 2);
    __bf16* Watt_bf   = (__bf16*)carve((size_t)H_DIM * E_DIM * 2);
    __bf16* Wed_bf    = (__bf16*)carve((size_t)H_DIM * E_DIM * 2);
    __bf16* Wout_bf   = (__bf16*)carve((size_t)D_DIM * 2 * H_DIM * 2);
    __bf16* prev_bf   = (__bf16*)carve((size_t)P_LEN * B_DIM * D_DIM * 2);
    __bf16* Ainit_bf  = (__bf16*)carve((size_t)B_DIM * E_DIM * 2);
    float*  hbuf      = (float*) carve((size_t)B_DIM * H_DIM * 4);
    __bf16* hbf       = (__bf16*)carve((size_t)B_DIM * H_DIM * 2);
    float*  gi        = (float*) carve((size_t)B_DIM * G3 * 4);
    float*  gh        = (float*) carve((size_t)B_DIM * G3 * 4);
    __bf16* concat_bf = (__bf16*)carve((size_t)B_DIM * 2 * H_DIM * 2);
    __bf16* x_bf      = (__bf16*)carve((size_t)B_DIM * D_DIM * 2);

    auto cvt = [&](const float* s, __bf16* d, int n) {
        cvt_f32_bf16<<<(n + 255) / 256, 256, 0, stream>>>(s, d, n);
    };
    cvt(W_ih, Wih_bf, G3 * D_DIM);
    cvt(W_hh, Whh_bf, G3 * H_DIM);
    cvt(W_att, Watt_bf, H_DIM * E_DIM);
    cvt(W_enc_dec, Wed_bf, H_DIM * E_DIM);
    cvt(W_out, Wout_bf, D_DIM * 2 * H_DIM);
    cvt(prev_poses, prev_bf, P_LEN * B_DIM * D_DIM);
    build_init_A<<<(B_DIM * E_DIM + 255) / 256, 256, 0, stream>>>(enc_hidden, Ainit_bf);

    // h0 = Ainit @ W_enc_dec^T + b_enc_dec  (M=256,N=512,K=512)
    gemm_bf16_wmma<<<dim3(H_DIM / 64, B_DIM / 64), 128, 0, stream>>>(
        Ainit_bf, E_DIM, Wed_bf, E_DIM, b_enc_dec,
        hbuf, H_DIM, hbf, H_DIM, B_DIM, H_DIM, E_DIM);

    // enc_proj (M=S*B, N=H, K=E), stored bf16 as ep[b][s][h]
    enc_proj_wmma<<<dim3(H_DIM / 64, (S_LEN * B_DIM) / 64), 128, 0, stream>>>(
        enc_states, Watt_bf, b_att, ep);

    dim3 ggrid(G3 / 64, B_DIM / 64);
    // warm-up GRU over previous poses
    for (int t = 0; t < P_LEN; ++t) {
        gru_gemm_wmma<<<ggrid, 128, 0, stream>>>(
            prev_bf + (size_t)t * B_DIM * D_DIM, hbf, Wih_bf, Whh_bf, gi, gh);
        gru_gates<<<(B_DIM * H_DIM) / 256, 256, 0, stream>>>(
            gi, gh, b_ih, b_hh, hbuf, hbf, concat_bf);
    }

    // decode loop
    for (int t = 0; t < T_STEPS; ++t) {
        const __bf16* x = (t == 0) ? (prev_bf + (size_t)(P_LEN - 1) * B_DIM * D_DIM) : x_bf;
        gru_gemm_wmma<<<ggrid, 128, 0, stream>>>(x, hbf, Wih_bf, Whh_bf, gi, gh);
        gru_gates<<<(B_DIM * H_DIM) / 256, 256, 0, stream>>>(
            gi, gh, b_ih, b_hh, hbuf, hbf, concat_bf);
        attention_kernel<<<B_DIM, 256, 0, stream>>>(ep, hbuf, concat_bf);
        // pose = [hc|ctx] @ W_out^T + b_out  (M=256,N=64,K=1024); also bf16 -> next x
        gemm_bf16_wmma<<<dim3(1, B_DIM / 64), 128, 0, stream>>>(
            concat_bf, 2 * H_DIM, Wout_bf, 2 * H_DIM, b_out,
            out + (size_t)t * B_DIM * D_DIM, D_DIM, x_bf, D_DIM,
            B_DIM, D_DIM, 2 * H_DIM);
    }
}
